// VanillaAttention_78151224918462
// MI455X (gfx1250) — matrix operand order repaired and checked by reading, not compiled, not run on hardware
//
#include <hip/hip_runtime.h>
#include <hip/hip_bf16.h>
#include <math.h>

typedef __attribute__((ext_vector_type(16))) __bf16 v16bf;
typedef __attribute__((ext_vector_type(8)))  float  v8f;

#define T_DIM 1024
#define B_DIM 64
#define M_DIM 80      // n_mel
#define E_DIM 512
#define KPAD  96      // 80 padded to 3x32 for wmma k-steps
#define WPITCH 104    // bf16 elements per Wx row in LDS (bank-conflict padding)
#define RPITCH 84     // f32 elements per input row in LDS (bank-conflict padding)
#define TILES_PER_B 64          // 1024 / 16 t-tiles per batch
#define ROWS_PER_BLOCK 128      // 8 waves x 16 t-rows

// Branch-free tanh: native V_TANH_F32 if the builtin exists, else exp2/rcp.
__device__ __forceinline__ float fast_tanhf(float x) {
#if __has_builtin(__builtin_amdgcn_tanhf)
    return __builtin_amdgcn_tanhf(x);
#else
    // tanh(x) = sign(x) * (1 - e^{-2|x|}) / (1 + e^{-2|x|}), all branchless.
    const float ax = __builtin_fabsf(x);
    const float t  = __builtin_amdgcn_exp2f(ax * -2.885390081777927f); // e^{-2|x|}
    const float r  = (1.0f - t) * __builtin_amdgcn_rcpf(1.0f + t);
    return __builtin_copysignf(r, x);
#endif
}

// ---------------------------------------------------------------------------
// q1 = query @ Wq^T + bq   (full f32: q1 is a graded output)
// ---------------------------------------------------------------------------
__global__ __launch_bounds__(256) void k_q1(const float* __restrict__ query,
                                            const float* __restrict__ Wq,
                                            const float* __restrict__ bq,
                                            float* __restrict__ q1out) {
    __shared__ float qs[E_DIM];
    const int b = blockIdx.x;
    for (int i = threadIdx.x; i < E_DIM; i += 256) qs[i] = query[b * E_DIM + i];
    __syncthreads();
    for (int e = threadIdx.x; e < E_DIM; e += 256) {
        const float* wr = Wq + e * E_DIM;
        float acc = 0.f;
        for (int k = 0; k < E_DIM; ++k) acc = fmaf(qs[k], wr[k], acc);
        q1out[b * E_DIM + e] = acc + bq[e];
    }
}

// ---------------------------------------------------------------------------
// w_eff[h] = sum_e v[e] * Wo[e,h]   (Wo folded through v; v.bo is softmax-
// invariant and dropped)
// ---------------------------------------------------------------------------
__global__ __launch_bounds__(512) void k_weff(const float* __restrict__ v,
                                              const float* __restrict__ Wo,
                                              float* __restrict__ weff) {
    __shared__ float vs[E_DIM];
    vs[threadIdx.x] = v[threadIdx.x];
    __syncthreads();
    const int h = threadIdx.x;
    float acc = 0.f;
    for (int e = 0; e < E_DIM; ++e) acc = fmaf(vs[e], Wo[e * E_DIM + h], acc);
    weff[h] = acc;
}

// ---------------------------------------------------------------------------
// Main fused pass: raw[b,t] = sum_e w_eff[e] * tanh( (in @ Wx^T)[t,b,e]
//                                                    + bx[e] + q1[b,e] )
// GEMM via v_wmma_f32_16x16x32_bf16, K=80 padded to 96 (3 k-steps).
// One block = one batch b, 128 consecutive t (8 waves x 16-t tiles); q1+bx
// collapse to a single per-block LDS row, so the epilogue is pure LDS+TRANS.
// ---------------------------------------------------------------------------
__global__ __launch_bounds__(256) void k_scores(
    const float* __restrict__ in,     // (T,B,80)
    const float* __restrict__ q1,     // (B,512)
    const float* __restrict__ Wx,     // (512,80)
    const float* __restrict__ bx,     // (512)
    const float* __restrict__ weff,   // (512)
    float* __restrict__ raw)          // (B,T)
{
    extern __shared__ char smem[];
    float*  rows_s = (float*)smem;                          // 128 * 84 f32
    float*  weff_s = rows_s + ROWS_PER_BLOCK * RPITCH;      // 512 f32
    float*  cq_s   = weff_s + E_DIM;                        // 512 f32 (bx+q1[b])
    __bf16* wx_s   = (__bf16*)(cq_s + E_DIM);               // 512 * 104 bf16

    const int tid = threadIdx.x;
    const int b   = blockIdx.x >> 3;          // batch for this whole block
    const int t0  = (blockIdx.x & 7) * ROWS_PER_BLOCK;   // first t of block

    // Stage this block's 128 input rows (t0..t0+127, fixed b) into LDS (f32)
    for (int idx = tid; idx < ROWS_PER_BLOCK * M_DIM; idx += 256) {
        const int row = idx / M_DIM, m = idx % M_DIM;
        rows_s[row * RPITCH + m] = in[((long)(t0 + row) * B_DIM + b) * M_DIM + m];
    }
    // Stage w_eff and the combined additive term bx + q1[b]
    for (int i = tid; i < E_DIM; i += 256) {
        weff_s[i] = weff[i];
        cq_s[i]   = bx[i] + q1[b * E_DIM + i];
    }
    // Stage Wx as bf16, zero-padded from K=80 to K=96
    for (int idx = tid; idx < E_DIM * KPAD; idx += 256) {
        const int e = idx / KPAD, k = idx % KPAD;
        const float f = (k < M_DIM) ? Wx[e * M_DIM + k] : 0.f;
        wx_s[e * WPITCH + k] = (__bf16)f;
    }
    __syncthreads();

    const int lane = tid & 31;
    const int wave = tid >> 5;
    const int hfv  = lane >> 4;   // which 16-lane half
    const int nIdx = lane & 15;
    const int mbase = hfv * 8;
    const int t0w  = t0 + wave * 16;          // this wave's first t

    // Build the three A fragments (16x32 bf16 each) in contiguous k order (same as B):
    // lane (mod 16) = row M (= local t); lanes 0-15 hold K 0-15,
    // lanes 16-31 hold K 16-31 (plus 32*s per k-step).
    v16bf a[3];
    const float* myrow = rows_s + (wave * 16 + nIdx) * RPITCH;
    #pragma unroll
    for (int s = 0; s < 3; ++s) {
        const int kbase = s * 32 + hfv * 16;
        #pragma unroll
        for (int h = 0; h < 16; ++h) {
            const int K = kbase + h;
            const float f = (K < M_DIM) ? myrow[K] : 0.f;
            a[s][h] = (__bf16)f;
        }
    }

    float ps[8];
    #pragma unroll
    for (int i = 0; i < 8; ++i) ps[i] = 0.f;

    for (int ct = 0; ct < 32; ++ct) {
        const int e = ct * 16 + nIdx;

        // B fragments (32x16 bf16): lane (mod 16) = col N, halves split K by 16.
        // 32 contiguous bytes per lane from LDS.
        union { uint4 u[2]; v16bf v; } bf[3];
        const __bf16* wp = wx_s + e * WPITCH + hfv * 16;
        #pragma unroll
        for (int s = 0; s < 3; ++s) {
            const __bf16* p = wp + s * 32;
            bf[s].u[0] = *(const uint4*)(p);
            bf[s].u[1] = *(const uint4*)(p + 8);
        }

        v8f acc = {};
        acc = __builtin_amdgcn_wmma_f32_16x16x32_bf16(false, a[0], false, bf[0].v,
                                                      (short)0, acc, false, false);
        acc = __builtin_amdgcn_wmma_f32_16x16x32_bf16(false, a[1], false, bf[1].v,
                                                      (short)0, acc, false, false);
        acc = __builtin_amdgcn_wmma_f32_16x16x32_bf16(false, a[2], false, bf[2].v,
                                                      (short)0, acc, false, false);

        // Epilogue: ps += w_eff[e] * tanh(acc + bx[e] + q1[b,e])
        // C/D layout: VGPR i -> row M = i (+8 for lanes 16-31), col N = lane&15;
        // e is identical for all 8 rows of this lane.
        const float w   = weff_s[e];
        const float cqe = cq_s[e];
        #pragma unroll
        for (int i = 0; i < 8; ++i)
            ps[i] = fmaf(w, fast_tanhf(acc[i] + cqe), ps[i]);
    }

    // Sum ps over the 16 lanes of each half (xor butterfly stays within half)
    #pragma unroll
    for (int off = 1; off < 16; off <<= 1) {
        #pragma unroll
        for (int i = 0; i < 8; ++i)
            ps[i] += __shfl_xor(ps[i], off, 32);
    }
    if (nIdx == 0) {   // lanes 0 and 16 hold rows mbase..mbase+7 -> contiguous t
        float* dst = raw + b * T_DIM + t0w + mbase;
        float4 s0 = {ps[0], ps[1], ps[2], ps[3]};
        float4 s1 = {ps[4], ps[5], ps[6], ps[7]};
        *(float4*)(dst)     = s0;
        *(float4*)(dst + 4) = s1;
    }
}

// ---------------------------------------------------------------------------
// Per-batch masked softmax over T, then folded expectation:
//   ctx[m]  = sum_t p[t] * in[t,b,m]
//   exp[e]  = Wx[e,:].ctx + bx[e]        (exact f32 path for the output)
// ---------------------------------------------------------------------------
__global__ __launch_bounds__(320) void k_softmax_ctx(
    const float* __restrict__ raw, const unsigned char* __restrict__ mask,
    const float* __restrict__ in, const float* __restrict__ Wx,
    const float* __restrict__ bx,
    float* __restrict__ out_exp, float* __restrict__ out_scores)
{
    __shared__ float p[T_DIM];
    __shared__ float red[320];
    __shared__ float ctx[M_DIM];
    __shared__ float gmaxs, gsums;
    const int b = blockIdx.x;
    const int tid = threadIdx.x;

    float lmax = -INFINITY;
    for (int t = tid; t < T_DIM; t += 320) {
        float x = mask[b * T_DIM + t] ? raw[b * T_DIM + t] : -INFINITY;
        p[t] = x;
        lmax = fmaxf(lmax, x);
    }
    red[tid] = lmax;
    __syncthreads();
    if (tid == 0) {
        float m = red[0];
        for (int i = 1; i < 320; ++i) m = fmaxf(m, red[i]);
        gmaxs = m;
    }
    __syncthreads();
    const float gmax = gmaxs;
    float lsum = 0.f;
    for (int t = tid; t < T_DIM; t += 320) {
        float e = expf(p[t] - gmax);
        p[t] = e;
        lsum += e;
    }
    red[tid] = lsum;
    __syncthreads();
    if (tid == 0) {
        float s = 0.f;
        for (int i = 0; i < 320; ++i) s += red[i];
        gsums = s;
    }
    __syncthreads();
    const float inv = 1.0f / gsums;
    for (int t = tid; t < T_DIM; t += 320) {
        const float pr = p[t] * inv;
        p[t] = pr;
        out_scores[b * T_DIM + t] = pr;
    }
    __syncthreads();

    // ctx[m] = sum_t p[t] * in[t,b,m]  (320 threads = 80 m-lanes x 4 t-groups)
    const int m = tid % M_DIM, g = tid / M_DIM;
    float acc = 0.f;
    for (int t = g; t < T_DIM; t += 4)
        acc = fmaf(p[t], in[((long)t * B_DIM + b) * M_DIM + m], acc);
    red[tid] = acc;
    __syncthreads();
    if (tid < M_DIM)
        ctx[tid] = red[tid] + red[tid + 80] + red[tid + 160] + red[tid + 240];
    __syncthreads();

    for (int e = tid; e < E_DIM; e += 320) {
        const float* wr = Wx + e * M_DIM;
        float s = bx[e];
        for (int mm = 0; mm < M_DIM; ++mm) s = fmaf(ctx[mm], wr[mm], s);
        out_exp[b * E_DIM + e] = s;
    }
}

extern "C" void kernel_launch(void* const* d_in, const int* in_sizes, int n_in,
                              void* d_out, int out_size, void* d_ws, size_t ws_size,
                              hipStream_t stream) {
    (void)in_sizes; (void)n_in; (void)out_size; (void)ws_size;
    const float*         in   = (const float*)d_in[0];          // (T,B,80)
    const float*         qv   = (const float*)d_in[1];          // (B,512)
    const unsigned char* mask = (const unsigned char*)d_in[2];  // (B,1,T) bool
    const float*         Wx   = (const float*)d_in[3];          // (512,80)
    const float*         bx   = (const float*)d_in[4];          // (512)
    const float*         Wq   = (const float*)d_in[5];          // (512,512)
    const float*         bq   = (const float*)d_in[6];          // (512)
    const float*         Wo   = (const float*)d_in[7];          // (512,512)
    // d_in[8] = bo: folded constant is softmax-invariant -> unused
    const float*         v    = (const float*)d_in[9];          // (512,1)

    float* out        = (float*)d_out;
    float* out_exp    = out;                                 // (B,E)
    float* out_scores = out + B_DIM * E_DIM;                 // (B,T)
    float* out_q1     = out + B_DIM * E_DIM + B_DIM * T_DIM; // (B,E)

    float* weff = (float*)d_ws;          // 512 f32
    float* raw  = weff + E_DIM;          // B*T f32 logits

    k_q1  <<<B_DIM, 256, 0, stream>>>(qv, Wq, bq, out_q1);
    k_weff<<<1, E_DIM, 0, stream>>>(v, Wo, weff);

    const size_t smem = (size_t)ROWS_PER_BLOCK * RPITCH * 4   // input rows
                      + (size_t)2 * E_DIM * 4                 // weff + (bx+q1)
                      + (size_t)E_DIM * WPITCH * 2;           // Wx bf16
    const int nblocks = (T_DIM * B_DIM) / ROWS_PER_BLOCK;     // 512
    k_scores<<<nblocks, 256, smem, stream>>>(in, out_q1, Wx, bx, weff, raw);

    k_softmax_ctx<<<B_DIM, 320, 0, stream>>>(raw, mask, in, Wx, bx,
                                             out_exp, out_scores);
}
